// AttRouter_82197084111013
// MI455X (gfx1250) — matrix operand order repaired and checked by reading, not compiled, not run on hardware
//
#include <hip/hip_runtime.h>

// ---------------------------------------------------------------------------
// AttRouter for MI455X (gfx1250, wave32, WMMA bf16 16x16x32, f32 accumulate)
// B=4, S=2048, E=16, D=256, A=64. One wave per token.
// Memory-bound design: 256 MiB of q/k activations @ 23.3 TB/s ~= 11.5 us floor;
// bf16 WMMA compute ~3-4 us, so conversions must be cheap -> hw v_cvt bf16.
// ---------------------------------------------------------------------------

#define B_   4
#define S_   2048
#define E_   16
#define D_   256
#define A_   64
#define NW   8                 // waves per block
#define NTOK (B_ * S_)         // 8192 tokens
#define WFRAG_ELEM (8 * 4 * 32 * 16)  // per weight matrix: kchunks*ntiles*lanes*elems (bf16)

typedef __attribute__((ext_vector_type(16))) __bf16 v16bf;
typedef __attribute__((ext_vector_type(8)))  float  v8f;

union BF16Frag {
    v16bf        v;
    unsigned int u[8];
    uint4        q[2];
};

// Load a 16xK(=32) A-fragment slice from a row-major f32 row pointer.
// A layout (ISA 7.12.2, 16-bit A 16x32): lane<16 -> K = hi*8 + [0..8) then 16+hi*8+[0..8).
// fptrunc f32->bf16 lowers to hardware v_cvt (RNE), no bit-twiddling.
__device__ __forceinline__ void load_afrag_f32(BF16Frag& f, const float* __restrict__ p) {
    const float4* a = (const float4*)p;          // K = +0..7
    const float4* b = (const float4*)(p + 16);   // K = +16..23
    float4 x0 = a[0], x1 = a[1], x2 = b[0], x3 = b[1];
    f.v[0]  = (__bf16)x0.x; f.v[1]  = (__bf16)x0.y;
    f.v[2]  = (__bf16)x0.z; f.v[3]  = (__bf16)x0.w;
    f.v[4]  = (__bf16)x1.x; f.v[5]  = (__bf16)x1.y;
    f.v[6]  = (__bf16)x1.z; f.v[7]  = (__bf16)x1.w;
    f.v[8]  = (__bf16)x2.x; f.v[9]  = (__bf16)x2.y;
    f.v[10] = (__bf16)x2.z; f.v[11] = (__bf16)x2.w;
    f.v[12] = (__bf16)x3.x; f.v[13] = (__bf16)x3.y;
    f.v[14] = (__bf16)x3.z; f.v[15] = (__bf16)x3.w;
}

__global__ __launch_bounds__(NW * 32) void AttRouter_82197084111013_kernel(
    const float* __restrict__ input_gate,   // (B,S,E,1)
    const float* __restrict__ qvec,         // (B,S,E,D)
    const float* __restrict__ kvec,         // (B,S,E,D)
    const float* __restrict__ Wq,           // (D,A)
    const float* __restrict__ Wk,           // (D,A)
    float* __restrict__ out)                // (B,S,E)
{
    extern __shared__ unsigned char smem[];
    __bf16* sWq = (__bf16*)smem;                       // 32 KiB
    __bf16* sWk = sWq + WFRAG_ELEM;                    // 32 KiB
    __bf16* sQ  = sWk + WFRAG_ELEM;                    // NW * 16x64 bf16
    __bf16* sK  = sQ + NW * (E_ * A_);
    float*  sSim = (float*)(sK + NW * (E_ * A_));      // NW * 16x16 f32
    float*  sG   = sSim + NW * (E_ * E_);              // NW * 16 f32

    const int tid = threadIdx.x;

    // ---- Stage Wq/Wk into LDS in WMMA B-fragment layout (bf16). ----
    // B layout (16-bit B 32x16): lane l -> column n = l&15; element e -> K = (l>>4)*8 + e (e<8), 16 + (l>>4)*8 + (e-8) (e>=8).
    for (int i = tid; i < WFRAG_ELEM; i += NW * 32) {
        int frag = i >> 9;            // 512 elements per fragment (32 lanes * 16)
        int rem  = i & 511;
        int lane = rem >> 4;
        int e    = rem & 15;
        int kc   = frag >> 2;
        int nt   = frag & 3;
        int col  = nt * 16 + (lane & 15);
        int d    = kc * 32 + (lane >> 4) * 8 + (e < 8 ? e : 16 + (e - 8));
        sWq[i] = (__bf16)Wq[d * A_ + col];
        sWk[i] = (__bf16)Wk[d * A_ + col];
    }
    __syncthreads();

    const int wv   = tid >> 5;
    const int lane = tid & 31;
    const int row  = lane & 15;   // M row (A frag) / N column (B,C frags)
    const int hi   = lane >> 4;
    const int tok  = blockIdx.x * NW + wv;

    const float* qrow = qvec + (size_t)tok * (E_ * D_) + row * D_;
    const float* krow = kvec + (size_t)tok * (E_ * D_) + row * D_;

    const v8f vzero = {0.f, 0.f, 0.f, 0.f, 0.f, 0.f, 0.f, 0.f};
    v8f qacc[4], kacc[4];
#pragma unroll
    for (int nt = 0; nt < 4; ++nt) { qacc[nt] = vzero; kacc[nt] = vzero; }

    // ---- Projection: (16x256) @ (256x64) for q and k, bf16 WMMA, f32 acc ----
#pragma unroll
    for (int kc = 0; kc < 8; ++kc) {
        const int d0 = kc * 32 + hi * 8;
        BF16Frag aq, ak;
        load_afrag_f32(aq, qrow + d0);
        load_afrag_f32(ak, krow + d0);
        if (kc < 7) {  // pull next K-chunk toward the caches
            __builtin_prefetch(qrow + d0 + 32, 0, 1);
            __builtin_prefetch(krow + d0 + 32, 0, 1);
        }
#pragma unroll
        for (int nt = 0; nt < 4; ++nt) {
            const int fo = (((kc << 2) | nt) * 32 + lane) * 16;  // bf16 elements
            BF16Frag bq, bk;
            const uint4* wp = (const uint4*)(sWq + fo);
            bq.q[0] = wp[0]; bq.q[1] = wp[1];
            const uint4* kp = (const uint4*)(sWk + fo);
            bk.q[0] = kp[0]; bk.q[1] = kp[1];
            qacc[nt] = __builtin_amdgcn_wmma_f32_16x16x32_bf16(
                false, aq.v, false, bq.v, (short)0, qacc[nt], false, false);
            kacc[nt] = __builtin_amdgcn_wmma_f32_16x16x32_bf16(
                false, ak.v, false, bk.v, (short)0, kacc[nt], false, false);
        }
    }

    // ---- Spill query/key tiles to LDS as row-major bf16 [16][64]. ----
    // C layout: VGPR r, lane<16 -> (M=r, N=lane); lane>=16 -> (M=r+8, N=lane-16).
    __bf16* q16 = sQ + wv * (E_ * A_);
    __bf16* k16 = sK + wv * (E_ * A_);
#pragma unroll
    for (int nt = 0; nt < 4; ++nt) {
#pragma unroll
        for (int r = 0; r < 8; ++r) {
            const int m = hi * 8 + r;
            const int a = nt * 16 + row;
            q16[m * A_ + a] = (__bf16)qacc[nt][r];
            k16[m * A_ + a] = (__bf16)kacc[nt][r];
        }
    }
    // Same-wave LDS ops are in order (DScnt) — no barrier needed for wave-private scratch.

    // ---- sim = Q @ K^T  (16x64 @ 64x16), two K=32 WMMA steps ----
    v8f sacc = vzero;
#pragma unroll
    for (int c = 0; c < 2; ++c) {
        BF16Frag aq, bk;
        // A-frag of Q: lane row = lane&15; K = hi*16+[0..16)
        const uint4* qa = (const uint4*)(q16 + row * A_ + c * 32 + hi * 16);
        const uint4* qb = (const uint4*)(q16 + row * A_ + c * 32 + hi * 16 + 8);
        aq.q[0] = qa[0]; aq.q[1] = qb[0];
        // B-frag of K^T == row-major K rows: lane column n = lane&15, K = hi*16+[0..16)
        const uint4* kb = (const uint4*)(k16 + row * A_ + c * 32 + hi * 16);
        bk.q[0] = kb[0]; bk.q[1] = kb[1];
        sacc = __builtin_amdgcn_wmma_f32_16x16x32_bf16(
            false, aq.v, false, bk.v, (short)0, sacc, false, false);
    }

    // ---- scale by 1/sqrt(64), park sim in LDS (C layout -> row major) ----
    float* simw = sSim + wv * (E_ * E_);
#pragma unroll
    for (int r = 0; r < 8; ++r)
        simw[(hi * 8 + r) * E_ + row] = sacc[r] * 0.125f;

    // ---- softmax over k, gate contraction, softmax over experts ----
    if (lane < 16) {
        const int m = lane;
        const float* gate = input_gate + (size_t)tok * E_;

        float mx = -3.402823466e+38f;
#pragma unroll
        for (int k = 0; k < E_; ++k) mx = fmaxf(mx, simw[m * E_ + k]);
        float se = 0.f, sg = 0.f;
#pragma unroll
        for (int k = 0; k < E_; ++k) {
            float e = __expf(simw[m * E_ + k] - mx);
            se += e;
            sg += e * gate[k];
        }
        float* g = sG + wv * E_;
        g[m] = sg / se;                      // gate_m = softmax(sim_m) . input_gate

        // second softmax across the 16 experts (values in wave-private LDS)
        float gm = g[m];
        float mx2 = -3.402823466e+38f;
#pragma unroll
        for (int k = 0; k < E_; ++k) mx2 = fmaxf(mx2, g[k]);
        float se2 = 0.f;
#pragma unroll
        for (int k = 0; k < E_; ++k) se2 += __expf(g[k] - mx2);
        out[(size_t)tok * E_ + m] = __expf(gm - mx2) / se2;
    }
}

extern "C" void kernel_launch(void* const* d_in, const int* in_sizes, int n_in,
                              void* d_out, int out_size, void* d_ws, size_t ws_size,
                              hipStream_t stream) {
    (void)in_sizes; (void)n_in; (void)out_size; (void)d_ws; (void)ws_size;
    const float* input_gate = (const float*)d_in[0];
    const float* qvec       = (const float*)d_in[1];
    const float* kvec       = (const float*)d_in[2];
    const float* Wq         = (const float*)d_in[3];
    const float* Wk         = (const float*)d_in[4];
    float*       out        = (float*)d_out;

    const size_t shmem = (size_t)(2 * WFRAG_ELEM * 2)                   // weights (64 KiB bf16)
                       + (size_t)NW * (2 * E_ * A_ * 2                  // q/k bf16 tiles
                                        + E_ * E_ * 4                   // sim f32
                                        + E_ * 4);                      // gate f32
    dim3 grid(NTOK / NW);
    dim3 block(NW * 32);
    AttRouter_82197084111013_kernel<<<grid, block, shmem, stream>>>(
        input_gate, qvec, kvec, Wq, Wk, out);
}
